// LoRAMultiHeadAttention_19756849561796
// MI455X (gfx1250) — matrix operand order repaired and checked by reading, not compiled, not run on hardware
//
#include <hip/hip_runtime.h>
#include <hip/hip_bf16.h>

// ---------------------------------------------------------------------------
// Types for CDNA5 WMMA (wave32): 16x16x32 bf16 -> f32 accumulate
// ---------------------------------------------------------------------------
typedef __attribute__((ext_vector_type(16))) __bf16 v16bf;
typedef __attribute__((ext_vector_type(8)))  __bf16 v8bf;
typedef __attribute__((ext_vector_type(8)))  float  v8f;
typedef __attribute__((ext_vector_type(4)))  unsigned int u32x4;
typedef __attribute__((ext_vector_type(8)))  int    i32x8;
typedef __attribute__((ext_vector_type(4)))  int    i32x4;

// ---------------------------------------------------------------------------
// Helpers
// ---------------------------------------------------------------------------
__device__ __forceinline__ __bf16 f2bf(float f) {
  unsigned int u = __builtin_bit_cast(unsigned int, f);
  // round-to-nearest-even to bf16
  unsigned int r = u + 0x7FFFu + ((u >> 16) & 1u);
  unsigned short s = (unsigned short)(r >> 16);
  return __builtin_bit_cast(__bf16, s);
}

__device__ __forceinline__ v16bf ld_frag(const __bf16* lo, const __bf16* hi) {
  v8bf a = *(const v8bf*)lo;
  v8bf b = *(const v8bf*)hi;
  return __builtin_shufflevector(a, b, 0, 1, 2, 3, 4, 5, 6, 7,
                                       8, 9, 10, 11, 12, 13, 14, 15);
}

__device__ __forceinline__ v8f wmma_bf16(v16bf a, v16bf b, v8f c) {
  // (neg_a, A, neg_b, B, c_mod, C, reuse_a, reuse_b)
  return __builtin_amdgcn_wmma_f32_16x16x32_bf16(false, a, false, b,
                                                 (short)0, c, false, false);
}

// ---------------------------------------------------------------------------
// Tensor Data Mover: DMA a 2-D tile (tile_h rows x tile_w bf16 elements,
// global row stride = stride_elems) into LDS, packed row-after-row.
// Tracked by TENSORcnt; EXEC is ignored by TDM so issue from one wave only
// (guarded by a uniform branch).
// ---------------------------------------------------------------------------
__device__ __forceinline__ unsigned int lds_addr_of(const void* p) {
  // Generic pointers to LDS carry the LDS byte offset in the low 32 bits.
  return (unsigned int)(unsigned long long)(size_t)p;
}

__device__ __forceinline__ void tdm_load_tile_2d(unsigned int lds_addr,
                                                 const void* gaddr,
                                                 unsigned int tile_w,
                                                 unsigned int tile_h,
                                                 unsigned long long stride_el) {
  unsigned long long ga = (unsigned long long)(size_t)gaddr;
  // D# group 0: count=1 | lds_addr | global_addr[56:0] | type=2 ("image")
  u32x4 g0 = (u32x4){
      1u,
      lds_addr,
      (unsigned int)ga,
      (unsigned int)((ga >> 32) & 0x01FFFFFFull) | 0x80000000u};
  // D# group 1: data_size=2B; tensor_dim0=tile_w; tensor_dim1=tile_h;
  // tile_dim0=tile_w; tile_dim1=tile_h; tile_dim2=0; dim0_stride=stride_el.
  i32x8 g1 = (i32x8){
      (int)(1u << 16),
      (int)((tile_w & 0xFFFFu) << 16),
      (int)(((tile_w >> 16) & 0xFFFFu) | ((tile_h & 0xFFFFu) << 16)),
      (int)(((tile_h >> 16) & 0xFFFFu) | ((tile_w & 0xFFFFu) << 16)),
      (int)(tile_h & 0xFFFFu),
      (int)(unsigned int)(stride_el & 0xFFFFFFFFull),
      (int)(unsigned int)((stride_el >> 32) & 0xFFFFull),
      0};
  i32x4 z4 = (i32x4){0, 0, 0, 0};
#if __has_include(<hip/amd_detail/amd_gfx1250_TDM.h>)
  // therock/clang-23 toolchain: 6-arg form
  i32x8 z8 = (i32x8){0, 0, 0, 0, 0, 0, 0, 0};
  __builtin_amdgcn_tensor_load_to_lds(g0, g1, z4, z4, z8, 0);
#else
  // ROCm 7.2 / clang-22: 5-arg form
  __builtin_amdgcn_tensor_load_to_lds(g0, g1, z4, z4, 0);
#endif
}

// ---------------------------------------------------------------------------
// Kernel: f32 -> bf16 elementwise (n divisible by 4)
// ---------------------------------------------------------------------------
__global__ void cvt_f32_bf16_kernel(const float* __restrict__ src,
                                    __bf16* __restrict__ dst, int n) {
  int i = (blockIdx.x * blockDim.x + threadIdx.x) * 4;
  if (i >= n) return;
  float4 v = *(const float4*)(src + i);
  dst[i + 0] = f2bf(v.x);
  dst[i + 1] = f2bf(v.y);
  dst[i + 2] = f2bf(v.z);
  dst[i + 3] = f2bf(v.w);
}

// ---------------------------------------------------------------------------
// Kernel: W_eff = W + scaling * Bm @ Am (rank-8 LoRA fold), cast to bf16.
// Bm: [N, 8], Am: [8, K]. Bm/Am may be null (plain convert).
// ---------------------------------------------------------------------------
__global__ void prep_weight_kernel(const float* __restrict__ W,
                                   const float* __restrict__ Bm,
                                   const float* __restrict__ Am,
                                   __bf16* __restrict__ out,
                                   int N, int K, float scaling) {
  int idx = blockIdx.x * blockDim.x + threadIdx.x;
  if (idx >= N * K) return;
  int n = idx / K;
  int k = idx - n * K;
  float w = W[idx];
  if (Bm != nullptr) {
    float acc = 0.0f;
#pragma unroll
    for (int r = 0; r < 8; ++r) acc += Bm[n * 8 + r] * Am[r * K + k];
    w += scaling * acc;
  }
  out[idx] = f2bf(w);
}

// ---------------------------------------------------------------------------
// Tiled WMMA GEMM: Cout[M,N] = A[M,K](bf16) @ Bw[N,K]^T(bf16) + bias[N]
// BM=128, BN=128, BK=32; 256 threads = 8 waves; wave tile 64x32 (4x2 WMMA).
// Tiles staged by the Tensor Data Mover into double-buffered LDS; wave 0
// issues the DMA for the next K-slab while all waves run WMMA on the
// current one (TENSORcnt + workgroup barrier for handoff).
// ---------------------------------------------------------------------------
template <bool BF16OUT>
__global__ __launch_bounds__(256)
void gemm_bias_kernel(const __bf16* __restrict__ A,
                      const __bf16* __restrict__ Bw,
                      const float* __restrict__ bias,
                      void* __restrict__ Cout,
                      int M, int N, int K) {
  __shared__ __align__(16) __bf16 As[2][128][32];
  __shared__ __align__(16) __bf16 Bs[2][128][32];

  const int bm = blockIdx.y * 128;
  const int bn = blockIdx.x * 128;
  const int tid = threadIdx.x;
  const int wave = tid >> 5;
  const int lane = tid & 31;
  const int l16 = lane & 15;
  const int lhalf = lane >> 4;
  const int wm = (wave & 1) * 64;
  const int wn = (wave >> 1) * 32;
  const bool issuer = (wave == 0);

  const __bf16* Ablk = A + (size_t)bm * K;
  const __bf16* Bblk = Bw + (size_t)bn * K;

  v8f acc[4][2];
#pragma unroll
  for (int i = 0; i < 4; ++i)
#pragma unroll
    for (int j = 0; j < 2; ++j)
      acc[i][j] = (v8f){0.f, 0.f, 0.f, 0.f, 0.f, 0.f, 0.f, 0.f};

  // Prologue: DMA first K-slab into buffer 0.
  if (issuer) {
    tdm_load_tile_2d(lds_addr_of(&As[0][0][0]), Ablk, 32u, 128u,
                     (unsigned long long)K);
    tdm_load_tile_2d(lds_addr_of(&Bs[0][0][0]), Bblk, 32u, 128u,
                     (unsigned long long)K);
  }

  int buf = 0;
  for (int k0 = 0; k0 < K; k0 += 32) {
    // Wait for current buffer's DMA, publish to all waves.
    if (issuer) __builtin_amdgcn_s_wait_tensorcnt(0);
    __syncthreads();

    // Kick off DMA of the next K-slab into the other buffer.
    if (issuer && (k0 + 32 < K)) {
      tdm_load_tile_2d(lds_addr_of(&As[buf ^ 1][0][0]), Ablk + k0 + 32, 32u,
                       128u, (unsigned long long)K);
      tdm_load_tile_2d(lds_addr_of(&Bs[buf ^ 1][0][0]), Bblk + k0 + 32, 32u,
                       128u, (unsigned long long)K);
    }

    // A fragments (16x32 layout: lane half 0 holds K {0..7,16..23}, half 1
    // holds K {8..15,24..31}); B columns == weight rows, same K pattern.
    v16bf afrag[4];
#pragma unroll
    for (int i = 0; i < 4; ++i) {
      const __bf16* r = &As[buf][wm + i * 16 + l16][0];
      afrag[i] = ld_frag(r + lhalf * 8, r + 16 + lhalf * 8);
    }
    v16bf bfrag[2];
#pragma unroll
    for (int j = 0; j < 2; ++j) {
      const __bf16* r = &Bs[buf][wn + j * 16 + l16][0];
      bfrag[j] = ld_frag(r + lhalf * 8, r + 16 + lhalf * 8);
    }
#pragma unroll
    for (int i = 0; i < 4; ++i)
#pragma unroll
      for (int j = 0; j < 2; ++j)
        acc[i][j] = wmma_bf16(afrag[i], bfrag[j], acc[i][j]);

    // All waves done reading this buffer before it can be DMA'd over.
    __syncthreads();
    buf ^= 1;
  }

  // Epilogue: C/D layout element (vgpr r, lane) -> m = r + 8*lhalf, n = l16.
#pragma unroll
  for (int i = 0; i < 4; ++i) {
#pragma unroll
    for (int j = 0; j < 2; ++j) {
      int ncol = bn + wn + j * 16 + l16;
      float bv = bias[ncol];
#pragma unroll
      for (int r = 0; r < 8; ++r) {
        int mr = bm + wm + i * 16 + r + 8 * lhalf;
        float val = acc[i][j][r] + bv;
        if (BF16OUT)
          ((__bf16*)Cout)[(size_t)mr * N + ncol] = f2bf(val);
        else
          ((float*)Cout)[(size_t)mr * N + ncol] = val;
      }
    }
  }
}

// ---------------------------------------------------------------------------
// Flash-style attention, bf16 WMMA, f32 online softmax.
// Grid: (T/64, B*H); 128 threads = 4 waves; each wave owns 16 query rows.
// q/k/v/o are [B, T, H*D] bf16; scores scaled by 1/sqrt(D).
// ---------------------------------------------------------------------------
__global__ __launch_bounds__(128)
void attention_kernel(const __bf16* __restrict__ q,
                      const __bf16* __restrict__ k,
                      const __bf16* __restrict__ v,
                      __bf16* __restrict__ o,
                      int T, int H, int Dh) {
  const int C = H * Dh;  // 1024
  const int bh = blockIdx.y;
  const int b = bh / H;
  const int h = bh - b * H;
  const int q0 = blockIdx.x * 64;
  const int tid = threadIdx.x;
  const int wave = tid >> 5;
  const int lane = tid & 31;
  const int l16 = lane & 15;
  const int lhalf = lane >> 4;

  __shared__ __align__(16) __bf16 Ks[32][64];   // keys:  [tk][d]
  __shared__ __align__(16) __bf16 VTs[64][32];  // V^T:   [d][tk]
  __shared__ __align__(16) __bf16 Ps[4][16][32];// per-wave P staging

  const size_t headoff = (size_t)b * T * C + (size_t)h * Dh;
  const __bf16* qb = q + headoff;
  const __bf16* kb = k + headoff;
  const __bf16* vb = v + headoff;
  __bf16* ob = o + headoff;

  // Resident Q fragments: rows q0 + wave*16 + l16, two K=32 slabs over d.
  const __bf16* qrow = qb + (size_t)(q0 + wave * 16 + l16) * C;
  v16bf qfrag[2];
#pragma unroll
  for (int f = 0; f < 2; ++f)
    qfrag[f] = ld_frag(qrow + f * 32 + lhalf * 8,
                       qrow + f * 32 + 16 + lhalf * 8);

  float mrow[8], lrow[8];
  v8f oacc[4];
#pragma unroll
  for (int r = 0; r < 8; ++r) { mrow[r] = -1e30f; lrow[r] = 0.0f; }
#pragma unroll
  for (int jd = 0; jd < 4; ++jd)
    oacc[jd] = (v8f){0.f, 0.f, 0.f, 0.f, 0.f, 0.f, 0.f, 0.f};

  const float sc = 0.125f;  // 1/sqrt(64)

  for (int kv0 = 0; kv0 < T; kv0 += 32) {
    // Stage K tile (32x64) with 16B chunks.
    for (int c = tid; c < 256; c += 128) {
      int row = c >> 3;
      int col = (c & 7) << 3;
      *(v8bf*)(&Ks[row][col]) =
          *(const v8bf*)(kb + (size_t)(kv0 + row) * C + col);
      if (kv0 + 32 < T)
        __builtin_prefetch(kb + (size_t)(kv0 + 32 + row) * C + col, 0, 0);
    }
    // Stage V transposed (scalar; writes spread over banks).
    for (int e = tid; e < 2048; e += 128) {
      int tk = e >> 6;
      int d = e & 63;
      VTs[d][tk] = vb[(size_t)(kv0 + tk) * C + d];
    }
    __syncthreads();

    // S = Q @ K^T : two 16-key n-tiles, accumulate over two d-slabs.
    v8f s[2];
    s[0] = (v8f){0.f, 0.f, 0.f, 0.f, 0.f, 0.f, 0.f, 0.f};
    s[1] = (v8f){0.f, 0.f, 0.f, 0.f, 0.f, 0.f, 0.f, 0.f};
#pragma unroll
    for (int j = 0; j < 2; ++j)
#pragma unroll
      for (int f = 0; f < 2; ++f) {
        const __bf16* r = &Ks[j * 16 + l16][0];
        v16bf bk = ld_frag(r + f * 32 + lhalf * 8,
                           r + f * 32 + 16 + lhalf * 8);
        s[j] = wmma_bf16(qfrag[f], bk, s[j]);
      }

    // Online softmax. Row m = r + 8*lhalf lives across the 16 lanes of this
    // half; xor-shuffle masks 1..8 reduce within each 16-lane half.
#pragma unroll
    for (int r = 0; r < 8; ++r) {
      float a0 = s[0][r] * sc;
      float a1 = s[1][r] * sc;
      float mx = fmaxf(a0, a1);
#pragma unroll
      for (int msk = 1; msk < 16; msk <<= 1)
        mx = fmaxf(mx, __shfl_xor(mx, msk, 32));
      float mnew = fmaxf(mrow[r], mx);
      float alpha = __expf(mrow[r] - mnew);
      float p0 = __expf(a0 - mnew);
      float p1 = __expf(a1 - mnew);
      float ps = p0 + p1;
#pragma unroll
      for (int msk = 1; msk < 16; msk <<= 1)
        ps += __shfl_xor(ps, msk, 32);
      lrow[r] = lrow[r] * alpha + ps;
      mrow[r] = mnew;
#pragma unroll
      for (int jd = 0; jd < 4; ++jd)
        oacc[jd][r] = oacc[jd][r] * alpha;
      // D-layout -> LDS so P can re-enter in A-fragment layout.
      Ps[wave][r + 8 * lhalf][l16] = f2bf(p0);
      Ps[wave][r + 8 * lhalf][16 + l16] = f2bf(p1);
    }

    // O += P(16x32) @ V(32x64); per-wave LDS, same-wave DS ordering via
    // hardware DScnt (compiler-inserted waits), no barrier needed.
    {
      const __bf16* pr = &Ps[wave][l16][0];
      v16bf pf = ld_frag(pr + lhalf * 8, pr + 16 + lhalf * 8);
#pragma unroll
      for (int jd = 0; jd < 4; ++jd) {
        const __bf16* vr = &VTs[jd * 16 + l16][0];
        v16bf bv = ld_frag(vr + lhalf * 8, vr + 16 + lhalf * 8);
        oacc[jd] = wmma_bf16(pf, bv, oacc[jd]);
      }
    }
    __syncthreads();
  }

  // Finalize: O /= l, write bf16.
#pragma unroll
  for (int r = 0; r < 8; ++r) {
    float inv = 1.0f / lrow[r];
    int t = q0 + wave * 16 + r + 8 * lhalf;
#pragma unroll
    for (int jd = 0; jd < 4; ++jd)
      ob[(size_t)t * C + jd * 16 + l16] = f2bf(oacc[jd][r] * inv);
  }
}

// ---------------------------------------------------------------------------
// Host-side launch
// ---------------------------------------------------------------------------
extern "C" void kernel_launch(void* const* d_in, const int* in_sizes, int n_in,
                              void* d_out, int out_size, void* d_ws,
                              size_t ws_size, hipStream_t stream) {
  (void)in_sizes; (void)n_in; (void)out_size; (void)ws_size;

  const float* x  = (const float*)d_in[0];
  const float* Wq = (const float*)d_in[1];
  const float* bq = (const float*)d_in[2];
  const float* Wk = (const float*)d_in[3];
  const float* bk = (const float*)d_in[4];
  const float* Wv = (const float*)d_in[5];
  const float* bv = (const float*)d_in[6];
  const float* Wo = (const float*)d_in[7];
  const float* bo = (const float*)d_in[8];
  const float* Aq = (const float*)d_in[9];
  const float* Bq = (const float*)d_in[10];
  const float* Av = (const float*)d_in[11];
  const float* Bv = (const float*)d_in[12];
  float* out = (float*)d_out;

  const int Bn = 4, T = 2048, C = 1024, H = 16, Dh = 64;
  const int M = Bn * T;             // 8192
  const float scaling = 32.0f / 8.0f;

  // Workspace layout (bf16 buffers), 256B aligned.
  char* ws = (char*)d_ws;
  size_t off = 0;
  auto take = [&](size_t bytes) -> char* {
    char* p = ws + off;
    off += (bytes + 255) & ~(size_t)255;
    return p;
  };
  __bf16* xb   = (__bf16*)take((size_t)M * C * 2);
  __bf16* wqe  = (__bf16*)take((size_t)C * C * 2);
  __bf16* wke  = (__bf16*)take((size_t)C * C * 2);
  __bf16* wve  = (__bf16*)take((size_t)C * C * 2);
  __bf16* woe  = (__bf16*)take((size_t)C * C * 2);
  __bf16* qbuf = (__bf16*)take((size_t)M * C * 2);
  __bf16* kbuf = (__bf16*)take((size_t)M * C * 2);
  __bf16* vbuf = (__bf16*)take((size_t)M * C * 2);
  __bf16* abuf = (__bf16*)take((size_t)M * C * 2);

  // 1) Precision conversion + LoRA weight fold.
  cvt_f32_bf16_kernel<<<dim3((M * C) / 4 / 256), 256, 0, stream>>>(x, xb, M * C);
  prep_weight_kernel<<<dim3((C * C) / 256), 256, 0, stream>>>(
      Wq, Bq, Aq, wqe, C, C, scaling);
  prep_weight_kernel<<<dim3((C * C) / 256), 256, 0, stream>>>(
      Wk, nullptr, nullptr, wke, C, C, scaling);
  prep_weight_kernel<<<dim3((C * C) / 256), 256, 0, stream>>>(
      Wv, Bv, Av, wve, C, C, scaling);
  prep_weight_kernel<<<dim3((C * C) / 256), 256, 0, stream>>>(
      Wo, nullptr, nullptr, woe, C, C, scaling);

  // 2) Q/K/V projections (bf16 out).
  dim3 ggrid(C / 128, M / 128);
  gemm_bias_kernel<true><<<ggrid, 256, 0, stream>>>(xb, wqe, bq, qbuf, M, C, C);
  gemm_bias_kernel<true><<<ggrid, 256, 0, stream>>>(xb, wke, bk, kbuf, M, C, C);
  gemm_bias_kernel<true><<<ggrid, 256, 0, stream>>>(xb, wve, bv, vbuf, M, C, C);

  // 3) Attention.
  attention_kernel<<<dim3(T / 64, Bn * H), 128, 0, stream>>>(
      qbuf, kbuf, vbuf, abuf, T, H, Dh);

  // 4) Output projection (f32 out).
  gemm_bias_kernel<false><<<ggrid, 256, 0, stream>>>(abuf, woe, bo, out, M, C, C);
}
